// SMN_72782515798267
// MI455X (gfx1250) — matrix operand order repaired and checked by reading, not compiled, not run on hardware
//
#include <hip/hip_runtime.h>
#include <hip/hip_bf16.h>
#include <math.h>

// ---------------------------------------------------------------------------
// SMN-style pipeline for gfx1250 (MI455X).
// WMMA path: __builtin_amdgcn_wmma_f32_16x16x32_f16 (probe-verified signature).
// Workspace requirement: ~575 MB of d_ws (float scratch), laid out below.
// ---------------------------------------------------------------------------

typedef __attribute__((ext_vector_type(16))) _Float16 v16h;
typedef __attribute__((ext_vector_type(8)))  float    v8f;

#define BB   1024
#define TT   32
#define CC   14
#define EE   200
#define HH   50
#define G3   150   // 3*H
#define OCC  8

// ---- packed f32->f16 LDS stores ------------------------------------------
__device__ __forceinline__ void cvt4_store(_Float16* dst, float4 v) {
  union { _Float16 h[4]; unsigned long long u; } p;
  p.h[0] = (_Float16)v.x; p.h[1] = (_Float16)v.y;
  p.h[2] = (_Float16)v.z; p.h[3] = (_Float16)v.w;
  *(unsigned long long*)dst = p.u;               // 8-byte LDS store
}
__device__ __forceinline__ void cvt2_store(_Float16* dst, float2 v) {
  union { _Float16 h[2]; unsigned int u; } p;
  p.h[0] = (_Float16)v.x; p.h[1] = (_Float16)v.y;
  *(unsigned int*)dst = p.u;                     // 4-byte LDS store
}

// ---- WMMA fragment loaders (wave32, 16x16x32 f16) -------------------------
// A (16x32, MxK): lanes 0-15 row M=lane, elems 0..15 -> K0..15;
// lanes 16-31 same rows, K16..31 (matches frag_b k order).
__device__ __forceinline__ v16h frag_a(const _Float16* p, int ldk, int lane) {
  const int row  = lane & 15;
  const int koff = (lane >> 4) << 4;
  const _Float16* s = p + row * ldk + koff;
  v16h a;
#pragma unroll
  for (int i = 0; i < 16; ++i) a[i] = s[i];
  return a;
}
// B (32x16, KxN): lanes 0-15 hold K=0..15 at col=lane; lanes 16-31 K=16..31.
// LDS holds W-tile row-major as [n][k] (i.e. B transposed), so this reads it.
__device__ __forceinline__ v16h frag_b(const _Float16* p, int ldk, int lane) {
  const int col  = lane & 15;
  const int koff = (lane >> 4) << 4;
  const _Float16* s = p + col * ldk + koff;
  v16h b;
#pragma unroll
  for (int i = 0; i < 16; ++i) b[i] = s[i];
  return b;
}

// ---- 1) fused (optional gather) GEMM: out[r][n] = A[r][:] . W[n][:] + bias[n]
// A row r = Asrc + (GATHER ? idx[r] : r) * K, length K (K % 4 == 0).
// Block: 128 thr / 4 waves; tile M=64 (16-row subtile per wave), N=16.
// Grid: (rows/64, ceil(N/16)).  Double-buffered LDS staging, 1 barrier/chunk.
template <int K, bool GATHER>
__global__ void __launch_bounds__(128)
xp_gemm_kernel(const float* __restrict__ Asrc, const int* __restrict__ idx,
               const float* __restrict__ W, const float* __restrict__ bias,
               float* __restrict__ out, int rows, int N, int ldo) {
  static_assert((K & 3) == 0, "K must be a multiple of 4");
  constexpr int NC = (K + 31) / 32;              // chunk count
  __shared__ _Float16 At[2][64 * 32];
  __shared__ _Float16 Bt[2][16 * 32];
  const int tid  = threadIdx.x;
  const int lane = tid & 31;
  const int wave = tid >> 5;
  const int r0 = blockIdx.x * 64;
  const int n0 = blockIdx.y * 16;

  // staging roles
  const int arow  = tid & 63;
  const int acol0 = (tid >> 6) << 4;             // 0 or 16 (uniform per wave)
  int rg = r0 + arow; if (rg >= rows) rg = rows - 1;
  const float* aptr = Asrc + (size_t)(GATHER ? idx[rg] : rg) * (size_t)K;

  const int brow  = tid >> 3;                    // 0..15
  const int bcol0 = (tid & 7) << 2;              // 0..28
  const int ng = n0 + brow;
  const bool wvalid = (ng < N);
  const float* wptr = W + (size_t)(wvalid ? ng : 0) * (size_t)K;

  auto stage = [&](int ci, int buf) {
    const int kc = ci * 32;
    if (kc + 32 < K) __builtin_prefetch(aptr + kc + 32 + acol0, 0, 3);
#pragma unroll
    for (int i = 0; i < 4; ++i) {
      const int k = kc + acol0 + 4 * i;          // float4-granular validity
      float4 v;
      if (k < K) v = *(const float4*)(aptr + k);
      else { v.x = v.y = v.z = v.w = 0.f; }
      cvt4_store(&At[buf][arow * 32 + acol0 + 4 * i], v);
    }
    const int kb = kc + bcol0;
    float4 wv;
    if (wvalid && kb < K) wv = *(const float4*)(wptr + kb);
    else { wv.x = wv.y = wv.z = wv.w = 0.f; }
    cvt4_store(&Bt[buf][brow * 32 + bcol0], wv);
  };

  v8f acc = {};
  stage(0, 0);
  __syncthreads();
#pragma unroll
  for (int ci = 0; ci < NC; ++ci) {
    const int buf = ci & 1;
    if (ci + 1 < NC) stage(ci + 1, buf ^ 1);     // overlap next-chunk staging
    v16h a = frag_a(At[buf] + wave * 16 * 32, 32, lane);
    v16h b = frag_b(Bt[buf], 32, lane);
    acc = __builtin_amdgcn_wmma_f32_16x16x32_f16(false, a, false, b,
                                                 (short)0, acc, false, false);
    __syncthreads();
  }

  const int ncol = n0 + (lane & 15);
  if (ncol < N) {
    const float bv = bias[ncol];
#pragma unroll
    for (int v = 0; v < 8; ++v) {
      int m = r0 + wave * 16 + v + ((lane >> 4) << 3);
      if (m < rows) out[(size_t)m * ldo + ncol] = acc[v] + bv;
    }
  }
}

// ---- 2) GRU scan: one thread per sequence, h state in LDS columns ---------
__global__ void __launch_bounds__(128)
gru_scan_kernel(const float* __restrict__ xp, const float* __restrict__ Whh,
                const float* __restrict__ bhh, float* __restrict__ hout,
                int S, int Tn) {
  __shared__ float sW[G3 * HH];
  __shared__ float sb[G3];
  __shared__ float hcur[HH * 128];
  __shared__ float hnew[HH * 128];
  const int tid = threadIdx.x;
  for (int i = tid; i < G3 * HH; i += 128) sW[i] = Whh[i];
  if (tid < G3) sb[tid] = bhh[tid];
#pragma unroll
  for (int k = 0; k < HH; ++k) hcur[k * 128 + tid] = 0.f;
  __syncthreads();
  const int seq = blockIdx.x * 128 + tid;
  if (seq >= S) return;
  for (int t = 0; t < Tn; ++t) {
    const float* xt = xp + ((size_t)seq * Tn + t) * G3;
    float* hw = hout + ((size_t)seq * Tn + t) * HH;
    for (int j = 0; j < HH; ++j) {
      float gr = sb[j], gz = sb[HH + j], gn = sb[2 * HH + j];
      const float* w_r = sW + j * HH;
      const float* w_z = sW + (HH + j) * HH;
      const float* w_n = sW + (2 * HH + j) * HH;
#pragma unroll 10
      for (int k = 0; k < HH; ++k) {
        float hk = hcur[k * 128 + tid];
        gr += w_r[k] * hk; gz += w_z[k] * hk; gn += w_n[k] * hk;
      }
      float r = 1.f / (1.f + __expf(-(xt[j] + gr)));
      float z = 1.f / (1.f + __expf(-(xt[HH + j] + gz)));
      float n = tanhf(xt[2 * HH + j] + r * gn);
      hnew[j * 128 + tid] = (1.f - z) * n + z * hcur[j * 128 + tid];
    }
    for (int j = 0; j < HH; ++j) {
      float hv = hnew[j * 128 + tid];
      hcur[j * 128 + tid] = hv;
      hw[j] = hv;
    }
  }
}

// ---- 3) ctx_hA = ctx_h @ A  (rows x 50 x 50) ------------------------------
__global__ void ctxhA_kernel(const float* __restrict__ ctx_h,
                             const float* __restrict__ A,
                             float* __restrict__ out, int rows) {
  __shared__ float sA[HH * HH];
  for (int i = threadIdx.x; i < HH * HH; i += blockDim.x) sA[i] = A[i];
  __syncthreads();
  int gid = blockIdx.x * blockDim.x + threadIdx.x;
  if (gid >= rows * HH) return;
  int r = gid / HH, k = gid % HH;
  const float* h = ctx_h + (size_t)r * HH;
  float acc = 0.f;
#pragma unroll 10
  for (int j = 0; j < HH; ++j) acc += h[j] * sA[j * HH + k];
  out[gid] = acc;
}

// ---- 4) matching: per (b,c) compute M1 (K=200) and M2 (K=50) via WMMA -----
// Stored in the reference's stack(axis=2)+reshape flat layout: (B,2,C,T,T).
__global__ void __launch_bounds__(128)
match_kernel(const int* __restrict__ ctx_idx, const int* __restrict__ cand_idx,
             const float* __restrict__ emb, const float* __restrict__ ctx_hA,
             const float* __restrict__ cand_h, float* __restrict__ M) {
  __shared__ _Float16 cE[32 * 224];
  __shared__ _Float16 rE[32 * 224];
  __shared__ _Float16 cA[32 * 64];
  __shared__ _Float16 rH[32 * 64];
  const int bc = blockIdx.x;
  const int b = bc / CC, c = bc % CC;
  const int tid = threadIdx.x, lane = tid & 31, wave = tid >> 5;

  // valid region: vectorized gather+convert (E = 200 = 50 float4 per row)
  for (int j = tid; j < 32 * 50; j += 128) {
    int row = j / 50, k4 = (j % 50) * 4;
    const float* cp = emb + (size_t)ctx_idx[b * TT + row] * EE + k4;
    const float* rp = emb + (size_t)cand_idx[(size_t)bc * TT + row] * EE + k4;
    cvt4_store(&cE[row * 224 + k4], *(const float4*)cp);
    cvt4_store(&rE[row * 224 + k4], *(const float4*)rp);
  }
  // zero pad cols 200..223 (24 halves = 6 u64 stores per row)
  for (int j = tid; j < 32 * 6; j += 128) {
    int row = j / 6, k4 = 200 + (j % 6) * 4;
    *(unsigned long long*)&cE[row * 224 + k4] = 0ull;
    *(unsigned long long*)&rE[row * 224 + k4] = 0ull;
  }
  // H = 50 = 25 float2 per row
  for (int j = tid; j < 32 * 25; j += 128) {
    int row = j / 25, k2 = (j % 25) * 2;
    const float* ap = ctx_hA + ((size_t)b * TT + row) * HH + k2;
    const float* hp = cand_h + ((size_t)bc * TT + row) * HH + k2;
    cvt2_store(&cA[row * 64 + k2], *(const float2*)ap);
    cvt2_store(&rH[row * 64 + k2], *(const float2*)hp);
  }
  // zero pad cols 50..63 (14 halves = 7 u32 stores per row)
  for (int j = tid; j < 32 * 7; j += 128) {
    int row = j / 7, k2 = 50 + (j % 7) * 2;
    *(unsigned int*)&cA[row * 64 + k2] = 0u;
    *(unsigned int*)&rH[row * 64 + k2] = 0u;
  }
  __syncthreads();

  const int tm = (wave >> 1) * 16, tn = (wave & 1) * 16;
  v8f a1 = {}, a2 = {};
#pragma unroll
  for (int kc = 0; kc < 224; kc += 32) {
    v16h af = frag_a(cE + tm * 224 + kc, 224, lane);
    v16h bf = frag_b(rE + tn * 224 + kc, 224, lane);
    a1 = __builtin_amdgcn_wmma_f32_16x16x32_f16(false, af, false, bf,
                                                (short)0, a1, false, false);
  }
#pragma unroll
  for (int kc = 0; kc < 64; kc += 32) {
    v16h af = frag_a(cA + tm * 64 + kc, 64, lane);
    v16h bf = frag_b(rH + tn * 64 + kc, 64, lane);
    a2 = __builtin_amdgcn_wmma_f32_16x16x32_f16(false, af, false, bf,
                                                (short)0, a2, false, false);
  }

  float* o1 = M + ((size_t)(b * 2 * CC + c)) * 1024;        // M1 slice
  float* o2 = M + ((size_t)(b * 2 * CC + CC + c)) * 1024;   // M2 slice
  const int n = tn + (lane & 15);
#pragma unroll
  for (int v = 0; v < 8; ++v) {
    int m = tm + v + ((lane >> 4) << 3);
    o1[m * 32 + n] = a1[v];
    o2[m * 32 + n] = a2[v];
  }
}

// ---- 5) conv 2->8 (3x3 VALID) + maxpool 3x3/3 -> feats (bc, 800) ----------
__global__ void conv_pool_kernel(const float* __restrict__ M,
                                 const float* __restrict__ conv_w,
                                 float* __restrict__ feats, int total) {
  __shared__ float sw[OCC * 2 * 9];
  if (threadIdx.x < OCC * 2 * 9) sw[threadIdx.x] = conv_w[threadIdx.x];
  __syncthreads();
  int gid = blockIdx.x * blockDim.x + threadIdx.x;
  if (gid >= total) return;
  int px = gid % 10, py = (gid / 10) % 10, oc = (gid / 100) % OCC;
  size_t bc = (size_t)gid / (100 * OCC);
  const float* m0 = M + bc * 2048;   // flat channel pair 2*bc, 2*bc+1
  const float* m1 = m0 + 1024;
  const float* w0 = sw + oc * 18;
  const float* w1 = w0 + 9;
  float best = -INFINITY;
#pragma unroll
  for (int dy = 0; dy < 3; ++dy)
#pragma unroll
    for (int dx = 0; dx < 3; ++dx) {
      int y = py * 3 + dy, x = px * 3 + dx;
      float acc = 0.f;
#pragma unroll
      for (int ky = 0; ky < 3; ++ky)
#pragma unroll
        for (int kx = 0; kx < 3; ++kx) {
          acc += m0[(y + ky) * 32 + (x + kx)] * w0[ky * 3 + kx]
               + m1[(y + ky) * 32 + (x + kx)] * w1[ky * 3 + kx];
        }
      best = fmaxf(best, acc);
    }
  feats[bc * 800 + oc * 100 + py * 10 + px] = best;
}

// ---- 6) head: fc1+relu, fc2, log_softmax, KL, mean ------------------------
__global__ void __launch_bounds__(128)
head_kernel(const float* __restrict__ Hs, const float* __restrict__ fc1_w,
            const float* __restrict__ fc1_b, const float* __restrict__ fc2_w,
            const float* __restrict__ fc2_b, const float* __restrict__ y_dev,
            float* __restrict__ out) {
  __shared__ float hid[100];
  __shared__ float lg[CC];
  const int b = blockIdx.x, tid = threadIdx.x;
  const float* flat = Hs + (size_t)b * (CC * HH);
  if (tid < 100) {
    float acc = fc1_b[tid];
    const float* w = fc1_w + (size_t)tid * (CC * HH);
    for (int k = 0; k < CC * HH; ++k) acc += flat[k] * w[k];
    hid[tid] = fmaxf(acc, 0.f);
  }
  __syncthreads();
  if (tid < CC) {
    float acc = fc2_b[tid];
    const float* w = fc2_w + (size_t)tid * 100;
#pragma unroll 10
    for (int k = 0; k < 100; ++k) acc += hid[k] * w[k];
    lg[tid] = acc;
  }
  __syncthreads();
  if (tid == 0) {
    float mx = lg[0];
    for (int i = 1; i < CC; ++i) mx = fmaxf(mx, lg[i]);
    float se = 0.f;
    for (int i = 0; i < CC; ++i) se += __expf(lg[i] - mx);
    float lse = mx + __logf(se);
    float kl = 0.f;
    const float* y = y_dev + (size_t)b * CC;
    for (int i = 0; i < CC; ++i) {
      float yi = y[i];
      if (yi > 0.f) kl += yi * (__logf(yi) - (lg[i] - lse));
    }
    atomicAdd(out, kl * (1.0f / (float)(BB * CC)));
  }
}

__global__ void zero_kernel(float* out) {
  if (threadIdx.x == 0 && blockIdx.x == 0) out[0] = 0.f;
}

// ---------------------------------------------------------------------------
extern "C" void kernel_launch(void* const* d_in, const int* in_sizes, int n_in,
                              void* d_out, int out_size, void* d_ws, size_t ws_size,
                              hipStream_t stream) {
  const int*   ctx_idx  = (const int*)d_in[0];
  const int*   cand_idx = (const int*)d_in[1];
  const float* y_dev    = (const float*)d_in[2];
  const float* emb      = (const float*)d_in[3];
  const float* A        = (const float*)d_in[4];
  const float* Wih_c    = (const float*)d_in[5];
  const float* Whh_c    = (const float*)d_in[6];
  const float* bih_c    = (const float*)d_in[7];
  const float* bhh_c    = (const float*)d_in[8];
  const float* Wih_r    = (const float*)d_in[9];
  const float* Whh_r    = (const float*)d_in[10];
  const float* bih_r    = (const float*)d_in[11];
  const float* bhh_r    = (const float*)d_in[12];
  const float* conv_w   = (const float*)d_in[13];
  const float* Wih2     = (const float*)d_in[14];
  const float* Whh2     = (const float*)d_in[15];
  const float* bih2     = (const float*)d_in[16];
  const float* bhh2     = (const float*)d_in[17];
  const float* fc1_w    = (const float*)d_in[18];
  const float* fc1_b    = (const float*)d_in[19];
  const float* fc2_w    = (const float*)d_in[20];
  const float* fc2_b    = (const float*)d_in[21];
  float* out = (float*)d_out;
  float* ws  = (float*)d_ws;

  // workspace layout (floats), total ~143.6M floats (~575 MB)
  size_t o = 0;
  float* xp_c  = ws + o; o += (size_t)BB * TT * G3;            // ctx xp
  float* xp_r  = ws + o; o += (size_t)BB * CC * TT * G3;       // cand xp
  float* ctxh  = ws + o; o += (size_t)BB * TT * HH;            // ctx GRU out
  float* candh = ws + o; o += (size_t)BB * CC * TT * HH;       // cand GRU out
  float* chA   = ws + o; o += (size_t)BB * TT * HH;            // ctx_h @ A
  float* Mbuf  = ws + o; o += (size_t)BB * 2 * CC * TT * TT;   // stacked M
  float* feats = ws + o; o += (size_t)BB * CC * 800;           // pooled feats
  float* xp2   = ws + o; o += (size_t)BB * CC * G3;            // GRU2 xp
  float* Hsb   = ws + o; o += (size_t)BB * CC * HH;            // GRU2 out

  zero_kernel<<<1, 32, 0, stream>>>(out);

  // xp = emb[idx] @ Wih^T + bih  (WMMA, gathered A, double-buffered staging)
  xp_gemm_kernel<EE, true><<<dim3((BB * TT) / 64, 10), 128, 0, stream>>>(
      emb, ctx_idx, Wih_c, bih_c, xp_c, BB * TT, G3, G3);
  xp_gemm_kernel<EE, true><<<dim3((BB * CC * TT) / 64, 10), 128, 0, stream>>>(
      emb, cand_idx, Wih_r, bih_r, xp_r, BB * CC * TT, G3, G3);

  // GRU scans
  gru_scan_kernel<<<BB / 128, 128, 0, stream>>>(xp_c, Whh_c, bhh_c, ctxh, BB, TT);
  gru_scan_kernel<<<(BB * CC) / 128, 128, 0, stream>>>(xp_r, Whh_r, bhh_r, candh,
                                                       BB * CC, TT);

  // ctx_h @ A
  ctxhA_kernel<<<(BB * TT * HH + 255) / 256, 256, 0, stream>>>(ctxh, A, chA, BB * TT);

  // matching matrices (WMMA)
  match_kernel<<<BB * CC, 128, 0, stream>>>(ctx_idx, cand_idx, emb, chA, candh, Mbuf);

  // conv + pool
  {
    int total = BB * CC * OCC * 100;
    conv_pool_kernel<<<(total + 255) / 256, 256, 0, stream>>>(Mbuf, conv_w, feats, total);
  }

  // GRU2 input projection (WMMA, dense A) + scan
  xp_gemm_kernel<800, false><<<dim3((BB * CC) / 64, 10), 128, 0, stream>>>(
      feats, (const int*)0, Wih2, bih2, xp2, BB * CC, G3, G3);
  gru_scan_kernel<<<BB / 128, 128, 0, stream>>>(xp2, Whh2, bhh2, Hsb, BB, CC);

  // head
  head_kernel<<<BB, 128, 0, stream>>>(Hsb, fc1_w, fc1_b, fc2_w, fc2_b, y_dev, out);
}
